// SelfAttention2D_52664888984187
// MI455X (gfx1250) — matrix operand order repaired and checked by reading, not compiled, not run on hardware
//
#include <hip/hip_runtime.h>
#include <hip/hip_bf16.h>

// ---------------------------------------------------------------------------
// SelfAttention2D on MI455X (gfx1250): WMMA f16->f32 everywhere.
// B=4, H=W=64 -> N=4096, C=512, F=64.
// ---------------------------------------------------------------------------

typedef _Float16 v16h __attribute__((ext_vector_type(16)));
typedef float    v8f  __attribute__((ext_vector_type(8)));

union Frag16 {            // 16 halves striped per-lane = one WMMA A/B operand
    v16h     h;
    uint32_t u[8];
};

#define WMMA_F16(A, B, C) \
    __builtin_amdgcn_wmma_f32_16x16x32_f16(false, (A), false, (B), (short)0, (C), false, false)

static constexpr int kB = 4;
static constexpr int kN = 4096;        // H*W
static constexpr int kC = 512;
static constexpr int kF = 64;
static constexpr int kRows = kB * kN;  // 16384

// ---------------------------------------------------------------------------
// Kernel 0: convert weights to f16 with WMMA-friendly (transposed) layouts.
//   WthT/WphT/WgT : [F, C]   (B-matrix for projection GEMM, K=C packed)
//   WoutT         : [C, F]   (B-matrix for output GEMM, K=F packed)
// ---------------------------------------------------------------------------
__global__ __launch_bounds__(256) void convert_weights_kernel(
    const float* __restrict__ Wth, const float* __restrict__ Wph,
    const float* __restrict__ Wg,  const float* __restrict__ Wout,
    _Float16* __restrict__ WthT, _Float16* __restrict__ WphT,
    _Float16* __restrict__ WgT,  _Float16* __restrict__ WoutT)
{
    int tid = blockIdx.x * 256 + threadIdx.x;   // 0 .. C*F-1
    int c = tid >> 6;                            // 0..511
    int f = tid & 63;                            // 0..63
    WthT[f * kC + c]  = (_Float16)Wth[c * kF + f];
    WphT[f * kC + c]  = (_Float16)Wph[c * kF + f];
    WgT [f * kC + c]  = (_Float16)Wg [c * kF + f];
    WoutT[c * kF + f] = (_Float16)Wout[f * kC + c];
}

// ---------------------------------------------------------------------------
// Kernel 1: fused theta/phi/g projections.  One wave = one 16(row) x 16(f)
// tile of all three outputs.  A fragment (x, f32->f16) is shared by 3 WMMAs.
//   theta, phi : row-major f16 [Rows, F]
//   gAct       : transposed f16 [B, F, N]  (B-matrix for P*G later)
// ---------------------------------------------------------------------------
__global__ __launch_bounds__(32) void proj_kernel(
    const float* __restrict__ x,
    const _Float16* __restrict__ WthT, const _Float16* __restrict__ WphT,
    const _Float16* __restrict__ WgT,
    const float* __restrict__ bth, const float* __restrict__ bph,
    const float* __restrict__ bg,
    _Float16* __restrict__ theta, _Float16* __restrict__ phi,
    _Float16* __restrict__ gAct)
{
    const int lane = threadIdx.x;
    const int m    = lane & 15;
    const bool hi  = lane >= 16;
    const int row  = blockIdx.x * 16 + m;       // global row (A-matrix M)
    const int f    = blockIdx.y * 16 + m;       // output feature (B-matrix N)

    const uint32_t* wt = (const uint32_t*)WthT; // [F][C] halves -> 256 u32/row
    const uint32_t* wp = (const uint32_t*)WphT;
    const uint32_t* wg = (const uint32_t*)WgT;

    v8f ct = {}, cp = {}, cg = {};

    for (int kc = 0; kc < kC / 32; ++kc) {
        const int k0 = kc * 32;
        // ---- A fragment: x[row, k0..k0+31], f32 -> f16, A 16x32 layout ----
        Frag16 a;
        const int ka = k0 + (hi ? 16 : 0);
        #pragma unroll
        for (int v = 0; v < 4; ++v) {
            float2 t0 = *(const float2*)(x + row * kC + ka + 2 * v);
            a.h[2 * v]     = (_Float16)t0.x;
            a.h[2 * v + 1] = (_Float16)t0.y;
            float2 t1 = *(const float2*)(x + row * kC + ka + 8 + 2 * v);
            a.h[8 + 2 * v]     = (_Float16)t1.x;
            a.h[8 + 2 * v + 1] = (_Float16)t1.y;
        }
        // ---- B fragments: W^T[f, k0..k0+31], B 32x16 layout ----
        const int kb = (k0 >> 1) + (hi ? 8 : 0);
        Frag16 bt, bp, bgf;
        #pragma unroll
        for (int v = 0; v < 8; ++v) {
            bt.u[v]  = wt[f * (kC / 2) + kb + v];
            bp.u[v]  = wp[f * (kC / 2) + kb + v];
            bgf.u[v] = wg[f * (kC / 2) + kb + v];
        }
        ct = WMMA_F16(a.h, bt.h,  ct);
        cp = WMMA_F16(a.h, bp.h,  cp);
        cg = WMMA_F16(a.h, bgf.h, cg);
    }

    const float vbt = bth[f], vbp = bph[f], vbg = bg[f];
    #pragma unroll
    for (int r = 0; r < 8; ++r) {
        const int orow = blockIdx.x * 16 + r + (hi ? 8 : 0);
        theta[orow * kF + f] = (_Float16)(ct[r] + vbt);
        phi[orow * kF + f]   = (_Float16)(cp[r] + vbp);
        const int b = orow >> 12;            // orow / 4096
        const int n = orow & (kN - 1);
        gAct[(b * kF + f) * kN + n] = (_Float16)(cg[r] + vbg);
    }
}

// ---------------------------------------------------------------------------
// Kernel 2: flash attention.  One wave = 16 query rows of one batch.
// Loop over 128 chunks of 32 keys:
//   S(16x32) = theta(16x64) @ phi^T  -> 4 WMMAs (two 16x16 C tiles, K=64)
//   online softmax (shfl_xor row reductions within half-wave + v_exp_f32)
//   P re-striped C-layout -> A-layout through LDS
//   O(16x64) += P(16x32) @ G(32x64)  -> 4 WMMAs
// ---------------------------------------------------------------------------
__global__ __launch_bounds__(32) void attn_kernel(
    const _Float16* __restrict__ theta, const _Float16* __restrict__ phi,
    const _Float16* __restrict__ gAct,  _Float16* __restrict__ attnOut)
{
    __shared__ _Float16 pbuf[16 * 32];          // P tile staging (1 KB)

    const int lane  = threadIdx.x;
    const int m     = lane & 15;
    const bool hi   = lane >= 16;
    const int batch = blockIdx.y;
    const int qtile = blockIdx.x;               // 0..255
    const int qrow  = batch * kN + qtile * 16 + m;

    const uint32_t* th32 = (const uint32_t*)theta;   // 32 u32 per row
    const uint32_t* ph32 = (const uint32_t*)phi;
    const uint32_t* g32  = (const uint32_t*)gAct;    // [B*F][N] -> 2048 u32/row

    // ---- preload theta A fragments (K = 0..31 and 32..63) ----
    Frag16 ta0, ta1;
    {
        const int ab = qrow * 32;
        const int ks = hi ? 8 : 0;
        #pragma unroll
        for (int v = 0; v < 4; ++v) {
            ta0.u[v]     = th32[ab + ks + v];
            ta0.u[4 + v] = th32[ab + 4 + ks + v];
            ta1.u[v]     = th32[ab + 16 + ks + v];
            ta1.u[4 + v] = th32[ab + 20 + ks + v];
        }
    }

    v8f o0 = {}, o1 = {}, o2 = {}, o3 = {};
    float mrow[8], lrow[8];
    #pragma unroll
    for (int r = 0; r < 8; ++r) { mrow[r] = -__builtin_inff(); lrow[r] = 0.0f; }

    for (int chunk = 0; chunk < kN / 32; ++chunk) {
        const int k0 = chunk * 32;

        // ---- phi B fragments: keys [k0..k0+15] and [k0+16..k0+31] ----
        const int key0 = batch * kN + k0 + m;
        const int fb   = hi ? 8 : 0;
        Frag16 p0a, p0b, p1a, p1b;
        #pragma unroll
        for (int v = 0; v < 8; ++v) {
            p0a.u[v] = ph32[key0 * 32 + fb + v];            // feats 0..31
            p0b.u[v] = ph32[key0 * 32 + 16 + fb + v];       // feats 32..63
            p1a.u[v] = ph32[(key0 + 16) * 32 + fb + v];
            p1b.u[v] = ph32[(key0 + 16) * 32 + 16 + fb + v];
        }
        // ---- G B fragments: K = 32 keys, N = 16 feats per tile ----
        const int kg = (k0 >> 1) + (hi ? 8 : 0);
        Frag16 gb0, gb1, gb2, gb3;
        #pragma unroll
        for (int v = 0; v < 8; ++v) {
            gb0.u[v] = g32[(batch * kF +  0 + m) * (kN / 2) + kg + v];
            gb1.u[v] = g32[(batch * kF + 16 + m) * (kN / 2) + kg + v];
            gb2.u[v] = g32[(batch * kF + 32 + m) * (kN / 2) + kg + v];
            gb3.u[v] = g32[(batch * kF + 48 + m) * (kN / 2) + kg + v];
        }

        // ---- scores S = theta @ phi^T ----
        v8f c0 = {}, c1 = {};
        c0 = WMMA_F16(ta0.h, p0a.h, c0);
        c0 = WMMA_F16(ta1.h, p0b.h, c0);
        c1 = WMMA_F16(ta0.h, p1a.h, c1);
        c1 = WMMA_F16(ta1.h, p1b.h, c1);

        // ---- online softmax (per-row, rows live across a 16-lane half) ----
        #pragma unroll
        for (int r = 0; r < 8; ++r) {
            float lm = fmaxf(c0[r], c1[r]);
            lm = fmaxf(lm, __shfl_xor(lm, 1));
            lm = fmaxf(lm, __shfl_xor(lm, 2));
            lm = fmaxf(lm, __shfl_xor(lm, 4));
            lm = fmaxf(lm, __shfl_xor(lm, 8));
            const float mn = fmaxf(mrow[r], lm);
            const float al = __expf(mrow[r] - mn);
            const float e0 = __expf(c0[r] - mn);
            const float e1 = __expf(c1[r] - mn);
            float rs = e0 + e1;
            rs += __shfl_xor(rs, 1);
            rs += __shfl_xor(rs, 2);
            rs += __shfl_xor(rs, 4);
            rs += __shfl_xor(rs, 8);
            lrow[r] = lrow[r] * al + rs;
            mrow[r] = mn;
            o0[r] *= al; o1[r] *= al; o2[r] *= al; o3[r] *= al;
            const int mr = r + (hi ? 8 : 0);
            pbuf[mr * 32 + m]      = (_Float16)e0;          // key column k0+m
            pbuf[mr * 32 + 16 + m] = (_Float16)e1;          // key column k0+16+m
        }
        __syncthreads();

        // ---- re-stripe P into A 16x32 layout ----
        Frag16 pa;
        {
            const uint32_t* pb32 = (const uint32_t*)pbuf;   // 16 u32 per row
            const int ps = m * 16 + (hi ? 8 : 0);
            #pragma unroll
            for (int v = 0; v < 4; ++v) {
                pa.u[v]     = pb32[ps + v];
                pa.u[4 + v] = pb32[ps + 4 + v];
            }
        }
        __syncthreads();

        // ---- O += P @ G ----
        o0 = WMMA_F16(pa.h, gb0.h, o0);
        o1 = WMMA_F16(pa.h, gb1.h, o1);
        o2 = WMMA_F16(pa.h, gb2.h, o2);
        o3 = WMMA_F16(pa.h, gb3.h, o3);
    }

    // ---- normalize and store attn_out (row-major f16 [Rows, F]) ----
    #pragma unroll
    for (int r = 0; r < 8; ++r) {
        const float inv = 1.0f / lrow[r];
        const int orow = batch * kN + qtile * 16 + r + (hi ? 8 : 0);
        attnOut[orow * kF +  0 + m] = (_Float16)(o0[r] * inv);
        attnOut[orow * kF + 16 + m] = (_Float16)(o1[r] * inv);
        attnOut[orow * kF + 32 + m] = (_Float16)(o2[r] * inv);
        attnOut[orow * kF + 48 + m] = (_Float16)(o3[r] * inv);
    }
}

// ---------------------------------------------------------------------------
// Kernel 3: out = attn_out @ W_out + b_out + x  (residual), f32 result.
// One wave = one 16x16 tile; K = F = 64 -> 2 WMMAs.
// ---------------------------------------------------------------------------
__global__ __launch_bounds__(32) void out_kernel(
    const _Float16* __restrict__ attnOut, const _Float16* __restrict__ WoutT,
    const float* __restrict__ bout, const float* __restrict__ x,
    float* __restrict__ out)
{
    const int lane = threadIdx.x;
    const int m    = lane & 15;
    const bool hi  = lane >= 16;
    const int row  = blockIdx.x * 16 + m;
    const int n    = blockIdx.y * 16 + m;          // output channel

    const uint32_t* a32 = (const uint32_t*)attnOut;  // 32 u32 per row
    const uint32_t* w32 = (const uint32_t*)WoutT;    // [C][F] -> 32 u32 per row

    Frag16 a0, a1;
    {
        const int ab = row * 32;
        const int ks = hi ? 8 : 0;
        #pragma unroll
        for (int v = 0; v < 4; ++v) {
            a0.u[v]     = a32[ab + ks + v];
            a0.u[4 + v] = a32[ab + 4 + ks + v];
            a1.u[v]     = a32[ab + 16 + ks + v];
            a1.u[4 + v] = a32[ab + 20 + ks + v];
        }
    }
    Frag16 b0, b1;
    {
        const int fb = hi ? 8 : 0;
        #pragma unroll
        for (int v = 0; v < 8; ++v) {
            b0.u[v] = w32[n * 32 + fb + v];        // K = 0..31
            b1.u[v] = w32[n * 32 + 16 + fb + v];   // K = 32..63
        }
    }

    v8f c = {};
    c = WMMA_F16(a0.h, b0.h, c);
    c = WMMA_F16(a1.h, b1.h, c);

    const float vb = bout[n];
    #pragma unroll
    for (int r = 0; r < 8; ++r) {
        const int orow = blockIdx.x * 16 + r + (hi ? 8 : 0);
        out[orow * kC + n] = c[r] + vb + x[orow * kC + n];
    }
}

// ---------------------------------------------------------------------------
// Host side
// ---------------------------------------------------------------------------
extern "C" void kernel_launch(void* const* d_in, const int* in_sizes, int n_in,
                              void* d_out, int out_size, void* d_ws, size_t ws_size,
                              hipStream_t stream)
{
    const float* x    = (const float*)d_in[0];
    const float* Wth  = (const float*)d_in[1];
    const float* bth  = (const float*)d_in[2];
    const float* Wph  = (const float*)d_in[3];
    const float* bph  = (const float*)d_in[4];
    const float* Wg   = (const float*)d_in[5];
    const float* bg   = (const float*)d_in[6];
    const float* Wout = (const float*)d_in[7];
    const float* bout = (const float*)d_in[8];

    // workspace layout (bytes)
    char* ws = (char*)d_ws;
    const size_t szW   = (size_t)kC * kF * sizeof(_Float16);       // 64 KB
    const size_t szAct = (size_t)kRows * kF * sizeof(_Float16);    // 2 MB
    size_t off = 0;
    _Float16* WthT  = (_Float16*)(ws + off); off += szW;
    _Float16* WphT  = (_Float16*)(ws + off); off += szW;
    _Float16* WgT   = (_Float16*)(ws + off); off += szW;
    _Float16* WoutT = (_Float16*)(ws + off); off += szW;
    _Float16* theta = (_Float16*)(ws + off); off += szAct;
    _Float16* phi   = (_Float16*)(ws + off); off += szAct;
    _Float16* gAct  = (_Float16*)(ws + off); off += szAct;
    _Float16* attnO = (_Float16*)(ws + off); off += szAct;
    if (ws_size < off) return;   // deterministic: never partially launch

    convert_weights_kernel<<<(kC * kF) / 256, 256, 0, stream>>>(
        Wth, Wph, Wg, Wout, WthT, WphT, WgT, WoutT);

    proj_kernel<<<dim3(kRows / 16, kF / 16), 32, 0, stream>>>(
        x, WthT, WphT, WgT, bth, bph, bg, theta, phi, gAct);

    attn_kernel<<<dim3(kN / 16, kB), 32, 0, stream>>>(
        theta, phi, gAct, attnO);

    out_kernel<<<dim3(kRows / 16, kC / 16), 32, 0, stream>>>(
        attnO, WoutT, bout, x, (float*)d_out);
}
